// Base_Encoder_46033459479309
// MI455X (gfx1250) — matrix operand order repaired and checked by reading, not compiled, not run on hardware
//
#include <hip/hip_runtime.h>

#define N_NODES 40000
#define N_EDGES 640000
#define DIM     128
#define DIM2    256
#define NLAYERS 3
#define NGRAPH  256
#define EPSV    1e-5f
#define SBF     264   // padded bf16 row stride for LDS h (spreads banks)

typedef __attribute__((ext_vector_type(16))) __bf16 v16bf;
typedef __attribute__((ext_vector_type(8)))  float  v8f;

#define WMMA_BF16(a, b, c) \
  __builtin_amdgcn_wmma_f32_16x16x32_bf16(false, (a), false, (b), (short)0, (c), false, false)

// ---- A-fragment (16x32 bf16, MxK) loaded from row-major fp32, converted ----
// ISA layout: lane L -> m = L&15, half = L>>4;
// elements 0..7  : K = half*8 + 0..7
// elements 8..15 : K = 16 + half*8 + 0..7
__device__ __forceinline__ v16bf load_a_f32(const float* A, int lda, int m, int k0, int half) {
  const float* p = A + (size_t)m * lda + k0 + half * 8;
  float4 f0 = *(const float4*)(p);
  float4 f1 = *(const float4*)(p + 4);
  float4 f2 = *(const float4*)(p + 16);
  float4 f3 = *(const float4*)(p + 20);
  v16bf a;
  a[0]=(__bf16)f0.x;  a[1]=(__bf16)f0.y;  a[2]=(__bf16)f0.z;  a[3]=(__bf16)f0.w;
  a[4]=(__bf16)f1.x;  a[5]=(__bf16)f1.y;  a[6]=(__bf16)f1.z;  a[7]=(__bf16)f1.w;
  a[8]=(__bf16)f2.x;  a[9]=(__bf16)f2.y;  a[10]=(__bf16)f2.z; a[11]=(__bf16)f2.w;
  a[12]=(__bf16)f3.x; a[13]=(__bf16)f3.y; a[14]=(__bf16)f3.z; a[15]=(__bf16)f3.w;
  return a;
}

// ---- A-fragment from LDS bf16 row-major (stride SBF), already bf16 ----
__device__ __forceinline__ v16bf load_a_lds(const __bf16* hbf, int m, int kt, int half) {
  union { v16bf v; uint4 q[2]; } u;
  const __bf16* p = hbf + m * SBF + kt * 32 + half * 8;
  u.q[0] = *(const uint4*)(p);
  u.q[1] = *(const uint4*)(p + 16);
  return u.v;
}

// ---------------- zero fill ----------------
__global__ void k_zero(float* p, int n) {
  int i = blockIdx.x * blockDim.x + threadIdx.x;
  if (i < n) p[i] = 0.0f;
}

// ---------------- weight -> bf16 B-fragment swizzle ----------------
// B (32x16, KxN) per ISA: lane L -> n = L&15; K = 16*(j>>3) + 8*(L>>4) + (j&7) for element j.
// Stored linear: ((kt*NT + nt)*32 + L)*16 + j
__global__ void k_prep_frag(const float* __restrict__ W, unsigned short* __restrict__ dst,
                            int Nout, int total) {
  int idx = blockIdx.x * blockDim.x + threadIdx.x;
  if (idx >= total) return;
  int j = idx & 15;
  int L = (idx >> 4) & 31;
  int t = idx >> 9;
  int NT = Nout >> 4;
  int kt = t / NT;
  int nt = t - kt * NT;
  int k = kt * 32 + ((j >> 3) << 4) + ((L >> 4) << 3) + (j & 7);
  int n = nt * 16 + (L & 15);
  __bf16 v = (__bf16)W[(size_t)k * Nout + n];
  dst[idx] = __builtin_bit_cast(unsigned short, v);
}

// ---------------- input projection: Y = X @ Wlin + b  (40000x128 @ 128x128) ----------------
// block = 8 waves; each wave owns one 16-row M-tile and all 8 N-tiles.
__global__ void k_gemm_lin(const float* __restrict__ X, const unsigned short* __restrict__ Wfrag,
                           const float* __restrict__ bias, float* __restrict__ Y) {
  int wave = threadIdx.x >> 5;
  int lane = threadIdx.x & 31;
  int mt = blockIdx.x * 8 + wave;
  if (mt >= N_NODES / 16) return;
  int m0 = mt * 16;
  int half = lane >> 4;
  const v16bf* B = (const v16bf*)Wfrag;

  v8f c[8];
#pragma unroll
  for (int nt = 0; nt < 8; ++nt) c[nt] = (v8f){0,0,0,0,0,0,0,0};

#pragma unroll
  for (int kt = 0; kt < 4; ++kt) {
    v16bf a = load_a_f32(X, DIM, m0 + (lane & 15), kt * 32, half);
#pragma unroll
    for (int nt = 0; nt < 8; ++nt) {
      v16bf b = B[(kt * 8 + nt) * 32 + lane];
      c[nt] = WMMA_BF16(a, b, c[nt]);
    }
  }
  int mrow = half * 8, col = lane & 15;
#pragma unroll
  for (int nt = 0; nt < 8; ++nt) {
#pragma unroll
    for (int r = 0; r < 8; ++r) {
      int row = m0 + mrow + r;
      int cc  = nt * 16 + col;
      Y[(size_t)row * DIM + cc] = c[nt][r] + bias[cc];
    }
  }
}

// ---------------- edge scatter: aggr[dst] += x[src] * w ----------------
// one wave32 per edge; 4 features per lane
__global__ void k_scatter(const float* __restrict__ x, const int* __restrict__ ei,
                          const float* __restrict__ ew, float* __restrict__ aggr) {
  int e = (blockIdx.x * blockDim.x + threadIdx.x) >> 5;
  int lane = threadIdx.x & 31;
  if (e >= N_EDGES) return;
  int s = ei[e];
  int d = ei[N_EDGES + e];
  float w = ew[e];
  const float4 v = *(const float4*)(x + (size_t)s * DIM + lane * 4);
  float* out = aggr + (size_t)d * DIM + lane * 4;
  atomicAdd(out + 0, v.x * w);
  atomicAdd(out + 1, v.y * w);
  atomicAdd(out + 2, v.z * w);
  atomicAdd(out + 3, v.w * w);
}

// ---------------- fused FFN per layer ----------------
// h = relu(LN(aggr@W1 + b1)); out = relu(h@W2 + b2 + aggr) -> h2out; BN partial sums.
// block = 256 threads (8 waves), one 16-row M-tile per block. h never leaves LDS.
__global__ void k_ffn(const float* __restrict__ aggr,
                      const unsigned short* __restrict__ W1frag, const float* __restrict__ bias1,
                      const float* __restrict__ lng, const float* __restrict__ lnb,
                      const unsigned short* __restrict__ W2frag, const float* __restrict__ bias2,
                      float* __restrict__ h2out, float* bnsum, float* bnsq) {
  __shared__ float hsm[16 * DIM2];
  __shared__ float red[16 * 16], red2[16 * 16];
  __shared__ float mu_s[16], rstd_s[16];
  __shared__ __attribute__((aligned(16))) __bf16 hbf[16 * SBF];

  int tid  = threadIdx.x;
  int wave = tid >> 5;
  int lane = tid & 31;
  int half = lane >> 4;
  int m0   = blockIdx.x * 16;
  const v16bf* B1 = (const v16bf*)W1frag;
  const v16bf* B2 = (const v16bf*)W2frag;

  // ---- GEMM1: wave handles n-tiles {2w, 2w+1} of the 256-wide h ----
  v8f c0 = (v8f){0,0,0,0,0,0,0,0};
  v8f c1 = (v8f){0,0,0,0,0,0,0,0};
#pragma unroll
  for (int kt = 0; kt < 4; ++kt) {
    v16bf a  = load_a_f32(aggr, DIM, m0 + (lane & 15), kt * 32, half);
    v16bf b0 = B1[(kt * 16 + 2 * wave + 0) * 32 + lane];
    v16bf b1 = B1[(kt * 16 + 2 * wave + 1) * 32 + lane];
    c0 = WMMA_BF16(a, b0, c0);
    c1 = WMMA_BF16(a, b1, c1);
  }
  int mrow = half * 8, col = lane & 15;
#pragma unroll
  for (int r = 0; r < 8; ++r) {
    int row = mrow + r;
    int ca = (2 * wave + 0) * 16 + col;
    int cb = (2 * wave + 1) * 16 + col;
    hsm[row * DIM2 + ca] = c0[r] + bias1[ca];
    hsm[row * DIM2 + cb] = c1[r] + bias1[cb];
  }
  __syncthreads();

  // ---- LayerNorm stats over 256 cols per row ----
  {
    int row = tid >> 4, part = tid & 15;
    float s = 0.f, s2 = 0.f;
#pragma unroll
    for (int i = 0; i < 16; ++i) {
      float v = hsm[row * DIM2 + part * 16 + i];
      s += v; s2 += v * v;
    }
    red[row * 16 + part] = s;
    red2[row * 16 + part] = s2;
  }
  __syncthreads();
  if (tid < 16) {
    float s = 0.f, s2 = 0.f;
#pragma unroll
    for (int i = 0; i < 16; ++i) { s += red[tid * 16 + i]; s2 += red2[tid * 16 + i]; }
    float mu  = s * (1.0f / DIM2);
    float var = s2 * (1.0f / DIM2) - mu * mu;
    mu_s[tid]   = mu;
    rstd_s[tid] = rsqrtf(var + EPSV);
  }
  __syncthreads();

  // ---- normalize + ReLU, repack as bf16 (A layout for GEMM2 is row-major contiguous) ----
  {
    int row = tid >> 4, part = tid & 15;
    float mu = mu_s[row], rs = rstd_s[row];
#pragma unroll
    for (int i = 0; i < 16; ++i) {
      int cc = part * 16 + i;
      float v = (hsm[row * DIM2 + cc] - mu) * rs * lng[cc] + lnb[cc];
      v = v > 0.f ? v : 0.f;
      hbf[row * SBF + cc] = (__bf16)v;
    }
  }
  __syncthreads();

  // ---- GEMM2: K=256 (8 kt), wave handles n-tile = wave ----
  v8f d = (v8f){0,0,0,0,0,0,0,0};
#pragma unroll
  for (int kt = 0; kt < 8; ++kt) {
    v16bf a = load_a_lds(hbf, lane & 15, kt, half);
    v16bf b = B2[(kt * 8 + wave) * 32 + lane];
    d = WMMA_BF16(a, b, d);
  }

  // ---- epilogue: +b2 +aggr (residual), ReLU, store, BN partial sums ----
  float s = 0.f, s2 = 0.f;
  int ccg = wave * 16 + col;
#pragma unroll
  for (int r = 0; r < 8; ++r) {
    int row = m0 + mrow + r;
    float v = d[r] + bias2[ccg] + aggr[(size_t)row * DIM + ccg];
    v = v > 0.f ? v : 0.f;
    h2out[(size_t)row * DIM + ccg] = v;
    s += v; s2 += v * v;
  }
  atomicAdd(&bnsum[ccg], s);
  atomicAdd(&bnsq[ccg], s2);
}

// ---------------- BatchNorm finalize + outputs + pooling ----------------
__global__ void k_bn_finalize(float* __restrict__ h, const float* __restrict__ bnsum,
                              const float* __restrict__ bnsq,
                              const float* __restrict__ gamma, const float* __restrict__ beta,
                              const int* __restrict__ batch,
                              float* __restrict__ xs_out, float* __restrict__ pool_out) {
  int idx = blockIdx.x * blockDim.x + threadIdx.x;
  if (idx >= N_NODES * DIM) return;
  int n = idx >> 7, c = idx & 127;
  float mean = bnsum[c] * (1.0f / N_NODES);
  float var  = bnsq[c] * (1.0f / N_NODES) - mean * mean;
  float rstd = rsqrtf(var + EPSV);
  float v = (h[idx] - mean) * rstd * gamma[c] + beta[c];
  h[idx] = v;  // in-place: becomes next layer's x
  xs_out[(size_t)n * (NLAYERS * DIM) + c] = v;
  atomicAdd(pool_out + (size_t)batch[n] * (NLAYERS * DIM) + c, v);
}

extern "C" void kernel_launch(void* const* d_in, const int* in_sizes, int n_in,
                              void* d_out, int out_size, void* d_ws, size_t ws_size,
                              hipStream_t stream) {
  const float* x_in  = (const float*)d_in[0];
  const int*   ei    = (const int*)d_in[1];
  const int*   batch = (const int*)d_in[2];
  const float* ew    = (const float*)d_in[3];
  const float* linW  = (const float*)d_in[4];
  const float* linb  = (const float*)d_in[5];
  const float* W1    = (const float*)d_in[6];
  const float* b1    = (const float*)d_in[7];
  const float* lng   = (const float*)d_in[8];
  const float* lnb   = (const float*)d_in[9];
  const float* W2    = (const float*)d_in[10];
  const float* b2    = (const float*)d_in[11];
  const float* bng   = (const float*)d_in[12];
  const float* bnb   = (const float*)d_in[13];

  // workspace layout
  float* xcur  = (float*)d_ws;                          // N*128 f32 (also h2, in place)
  float* aggr  = xcur + (size_t)N_NODES * DIM;          // N*128 f32
  float* stats = aggr + (size_t)N_NODES * DIM;          // L*2*128 f32
  unsigned short* linfrag = (unsigned short*)(stats + NLAYERS * 2 * DIM);
  unsigned short* w1frag  = linfrag + 16384;            // 3 x (4*16*512)
  unsigned short* w2frag  = w1frag + 3 * 32768;         // 3 x (8*8*512)

  float* pool = (float*)d_out;                          // [G, 3*128]
  float* xs   = pool + (size_t)NGRAPH * NLAYERS * DIM;  // [N, 3*128]

  // zero pooled output + BN stats (d_out/d_ws are poisoned by harness)
  k_zero<<<(NGRAPH * NLAYERS * DIM + 255) / 256, 256, 0, stream>>>(pool, NGRAPH * NLAYERS * DIM);
  k_zero<<<3, 256, 0, stream>>>(stats, NLAYERS * 2 * DIM);

  // weight conversion + fragment swizzle (tiny, once per launch)
  k_prep_frag<<<64, 256, 0, stream>>>(linW, linfrag, DIM, 16384);
  for (int i = 0; i < NLAYERS; ++i) {
    k_prep_frag<<<128, 256, 0, stream>>>(W1 + (size_t)i * DIM * DIM2, w1frag + (size_t)i * 32768, DIM2, 32768);
    k_prep_frag<<<128, 256, 0, stream>>>(W2 + (size_t)i * DIM2 * DIM, w2frag + (size_t)i * 32768, DIM, 32768);
  }

  // input projection
  k_gemm_lin<<<(N_NODES / 16 + 7) / 8, 256, 0, stream>>>(x_in, linfrag, linb, xcur);

  for (int i = 0; i < NLAYERS; ++i) {
    k_zero<<<(N_NODES * DIM + 255) / 256, 256, 0, stream>>>(aggr, N_NODES * DIM);
    k_scatter<<<N_EDGES / 8, 256, 0, stream>>>(xcur, ei, ew, aggr);
    k_ffn<<<N_NODES / 16, 256, 0, stream>>>(aggr,
        w1frag + (size_t)i * 32768, b1 + i * DIM2, lng + i * DIM2, lnb + i * DIM2,
        w2frag + (size_t)i * 32768, b2 + i * DIM,
        xcur, stats + i * 2 * DIM, stats + i * 2 * DIM + DIM);
    k_bn_finalize<<<(N_NODES * DIM + 255) / 256, 256, 0, stream>>>(xcur,
        stats + i * 2 * DIM, stats + i * 2 * DIM + DIM,
        bng + i * DIM, bnb + i * DIM, batch, xs + i * DIM, pool + i * DIM);
  }
}
